// CausalSelfAttention_27144193311232
// MI455X (gfx1250) — matrix operand order repaired and checked by reading, not compiled, not run on hardware
//
#include <hip/hip_runtime.h>
#include <hip/hip_bf16.h>

typedef __attribute__((ext_vector_type(16))) _Float16 v16h;
typedef __attribute__((ext_vector_type(8)))  _Float16 v8h;
typedef __attribute__((ext_vector_type(4)))  _Float16 v4h;
typedef __attribute__((ext_vector_type(2)))  _Float16 v2h;
typedef __attribute__((ext_vector_type(8)))  float    v8f;
typedef __attribute__((ext_vector_type(4)))  int      v4i;

#define DMODEL 1024
#define NHEADS 16
#define DEPTH  64
#define BATCH  4
#define SEQ    2048
#define NTOK   (BATCH * SEQ)          // 8192
#define HTOK   ((size_t)SEQ * DEPTH)  // per (b,h) plane: 131072 halves

static __device__ __forceinline__ v8f wmma_f16(v16h a, v16h b, v8f c) {
    return __builtin_amdgcn_wmma_f32_16x16x32_f16(false, a, false, b, (short)0, c, false, false);
}
static __device__ __forceinline__ v16h cat16(v8h lo, v8h hi) {
    v16h r;
#pragma unroll
    for (int i = 0; i < 8; ++i) { r[i] = lo[i]; r[8 + i] = hi[i]; }
    return r;
}

// 16B async DMA global -> LDS (ASYNCcnt-tracked).
static __device__ __forceinline__ void cp_async16(void* lds, const void* g) {
#if __has_builtin(__builtin_amdgcn_global_load_async_to_lds_b128)
    typedef __attribute__((address_space(1))) v4i* gp_t;
    typedef __attribute__((address_space(3))) v4i* lp_t;
    __builtin_amdgcn_global_load_async_to_lds_b128(
        (gp_t)(unsigned long long)g,
        (lp_t)(unsigned)(unsigned long long)lds, 0, 0);
#else
    *(v8h*)lds = *(const v8h*)g;
#endif
}
static __device__ __forceinline__ void wait_async0() {
#if __has_builtin(__builtin_amdgcn_s_wait_asynccnt)
    __builtin_amdgcn_s_wait_asynccnt(0);
#else
    asm volatile("s_wait_asynccnt 0" ::: "memory");
#endif
}

// ---------------------------------------------------------------------------
// One-time weight prep: Wt[col][k] = (f16) W[k][col]  (1024x1024).
// ---------------------------------------------------------------------------
__global__ __launch_bounds__(256) void w_transpose_f16(const float* __restrict__ W,
                                                       _Float16* __restrict__ Wt) {
    __shared__ float t[32][33];
    const int tx = threadIdx.x & 31;
    const int ty = threadIdx.x >> 5;  // 0..7
    const int x0 = blockIdx.x * 32;
    const int y0 = blockIdx.y * 32;
#pragma unroll
    for (int i = 0; i < 4; ++i)
        t[ty + 8 * i][tx] = W[(size_t)(y0 + ty + 8 * i) * DMODEL + x0 + tx];
    __syncthreads();
#pragma unroll
    for (int i = 0; i < 4; ++i)
        Wt[(size_t)(x0 + ty + 8 * i) * DMODEL + y0 + tx] = (_Float16)t[tx][ty + 8 * i];
}

// ---------------------------------------------------------------------------
// Projection GEMM: Y = X[8192x1024](f32) * W + bias -> f16, W given as
// pre-transposed f16 Wt[col][k]. MODE 0: Q (scaled 1/8, [B,H,S,64]);
// 1: K ([B,H,S,64]); 2: V^T ([B,H,64,S], stores staged via LDS).
// Tile 64x128x64, 8 waves. Double-buffered: async B DMA + A conversion for
// tile i+1 overlap the WMMA burst on tile i; one barrier per k-step.
// ---------------------------------------------------------------------------
template <int MODE>
__global__ __launch_bounds__(256) void proj_qkv(const float* __restrict__ X,
                                                const _Float16* __restrict__ Wt,
                                                const float* __restrict__ bias,
                                                _Float16* __restrict__ out) {
    __shared__ _Float16 As[2][64][72];   // [buf][row][k]
    __shared__ _Float16 Bt[2][128][72];  // [buf][col][k]
    const int tid  = threadIdx.x;
    const int lane = tid & 31;
    const int wave = tid >> 5;
    const int wm   = wave & 3;
    const int wn   = wave >> 2;
    const int rowBase = blockIdx.x * 64;
    const int colBase = blockIdx.y * 128;
    const int n0   = lane & 15;
    const int half = lane >> 4;
    const int koff = half ? 8 : 0;
    const int kb   = half << 4;

    auto fillB = [&](int bf, int k0) {
#pragma unroll
        for (int i = 0; i < 4; ++i) {
            int vi = tid + i * 256;  // 1024 x 16B chunks
            int col = vi >> 3, kc = (vi & 7) * 8;
            cp_async16(&Bt[bf][col][kc], &Wt[(size_t)(colBase + col) * DMODEL + k0 + kc]);
        }
    };
    auto fillA = [&](int bf, int k0) {
#pragma unroll
        for (int i = 0; i < 4; ++i) {
            int vi = tid + i * 256;  // 1024 float4's
            int r = vi >> 4, c4 = (vi & 15) * 4;
            const float4 x = *(const float4*)&X[(size_t)(rowBase + r) * DMODEL + k0 + c4];
            v4h p = {(_Float16)x.x, (_Float16)x.y, (_Float16)x.z, (_Float16)x.w};
            *(v4h*)&As[bf][r][c4] = p;
        }
    };

    v8f acc[4];
#pragma unroll
    for (int c = 0; c < 4; ++c) acc[c] = (v8f){0, 0, 0, 0, 0, 0, 0, 0};

    fillB(0, 0);
    fillA(0, 0);
    wait_async0();
    __syncthreads();

    int buf = 0;
    for (int k0 = 0; k0 < DMODEL; k0 += 64) {
        const int nb = buf ^ 1;
        if (k0 + 64 < DMODEL) {
            fillB(nb, k0 + 64);          // async DMA overlaps WMMAs below
            fillA(nb, k0 + 64);
            if (k0 + 128 < DMODEL)
                __builtin_prefetch(&X[(size_t)(rowBase + (tid >> 4)) * DMODEL + k0 + 128 + (tid & 15) * 4], 0, 0);
        }

        // Hoist all fragments, then one WMMA burst.
        v16h afr[2], bfr[2][4];
#pragma unroll
        for (int f = 0; f < 2; ++f) {
            const _Float16* ar = &As[buf][wm * 16 + n0][f * 32];
            afr[f] = cat16(*(const v8h*)(ar + kb), *(const v8h*)(ar + kb + 8));
#pragma unroll
            for (int c = 0; c < 4; ++c) {
                const _Float16* br = &Bt[buf][wn * 64 + c * 16 + n0][f * 32 + kb];
                bfr[f][c] = cat16(*(const v8h*)br, *(const v8h*)(br + 8));
            }
        }
#pragma unroll
        for (int f = 0; f < 2; ++f)
#pragma unroll
            for (int c = 0; c < 4; ++c)
                acc[c] = wmma_f16(afr[f], bfr[f][c], acc[c]);

        wait_async0();
        __syncthreads();
        buf = nb;
    }

    if (MODE == 2) {
        // Stage [feat][tok] in LDS (Bt free now), then coalesced V^T rows.
        _Float16(*Ct)[72] = (_Float16(*)[72])Bt[0];
#pragma unroll
        for (int c = 0; c < 4; ++c) {
            const int fl = wn * 64 + c * 16 + n0;
            const float bv = bias[colBase + fl];
#pragma unroll
            for (int r = 0; r < 8; ++r)
                Ct[fl][wm * 16 + r + 8 * half] = (_Float16)(acc[c][r] + bv);
        }
        __syncthreads();
        const int fl   = tid >> 1;            // 0..127
        const int th   = (tid & 1) * 32;      // token offset within tile
        const int feat = colBase + fl;
        const int h2 = feat >> 6, dep = feat & 63;
        const int bb = rowBase >> 11;
        const int sposB = (rowBase & 2047) + th;
        _Float16* dst = out + ((size_t)(bb * NHEADS + h2) * DEPTH + dep) * SEQ + sposB;
#pragma unroll
        for (int j = 0; j < 4; ++j)
            *(v8h*)(dst + j * 8) = *(const v8h*)&Ct[fl][th + j * 8];
    } else {
#pragma unroll
        for (int c = 0; c < 4; ++c) {
            const int feat = colBase + wn * 64 + c * 16 + n0;
            const int h = feat >> 6, dep = feat & 63;
            const float bv = bias[feat];
#pragma unroll
            for (int r = 0; r < 8; ++r) {
                const int tok  = rowBase + wm * 16 + r + 8 * half;
                const int b    = tok >> 11, spos = tok & 2047;
                float v = acc[c][r] + bv;
                if (MODE == 0) v *= 0.125f;  // 1/sqrt(depth) folded into Q
                out[((size_t)(b * NHEADS + h) * SEQ + spos) * DEPTH + dep] = (_Float16)v;
            }
        }
    }
}

// ---------------------------------------------------------------------------
// Output projection: d_out = Oh[8192x1024](f16) * Wo + bias -> f32.
// Fully async double-buffered tiles (A is already f16).
// ---------------------------------------------------------------------------
__global__ __launch_bounds__(256) void proj_out(const _Float16* __restrict__ X,
                                                const _Float16* __restrict__ Wt,
                                                const float* __restrict__ bias,
                                                float* __restrict__ out) {
    __shared__ _Float16 As[2][64][72];
    __shared__ _Float16 Bt[2][128][72];
    const int tid  = threadIdx.x;
    const int lane = tid & 31;
    const int wave = tid >> 5;
    const int wm   = wave & 3;
    const int wn   = wave >> 2;
    const int rowBase = blockIdx.x * 64;
    const int colBase = blockIdx.y * 128;
    const int n0   = lane & 15;
    const int half = lane >> 4;
    const int koff = half ? 8 : 0;
    const int kb   = half << 4;

    auto fillTiles = [&](int bf, int k0) {
#pragma unroll
        for (int i = 0; i < 4; ++i) {
            int vi = tid + i * 256;
            int col = vi >> 3, kc = (vi & 7) * 8;
            cp_async16(&Bt[bf][col][kc], &Wt[(size_t)(colBase + col) * DMODEL + k0 + kc]);
        }
#pragma unroll
        for (int i = 0; i < 2; ++i) {
            int vi = tid + i * 256;  // 512 x 16B chunks
            int r = vi >> 3, kc = (vi & 7) * 8;
            cp_async16(&As[bf][r][kc], &X[(size_t)(rowBase + r) * DMODEL + k0 + kc]);
        }
    };

    v8f acc[4];
#pragma unroll
    for (int c = 0; c < 4; ++c) acc[c] = (v8f){0, 0, 0, 0, 0, 0, 0, 0};

    fillTiles(0, 0);
    wait_async0();
    __syncthreads();

    int buf = 0;
    for (int k0 = 0; k0 < DMODEL; k0 += 64) {
        const int nb = buf ^ 1;
        if (k0 + 64 < DMODEL) fillTiles(nb, k0 + 64);

        v16h afr[2], bfr[2][4];
#pragma unroll
        for (int f = 0; f < 2; ++f) {
            const _Float16* ar = &As[buf][wm * 16 + n0][f * 32];
            afr[f] = cat16(*(const v8h*)(ar + kb), *(const v8h*)(ar + kb + 8));
#pragma unroll
            for (int c = 0; c < 4; ++c) {
                const _Float16* br = &Bt[buf][wn * 64 + c * 16 + n0][f * 32 + kb];
                bfr[f][c] = cat16(*(const v8h*)br, *(const v8h*)(br + 8));
            }
        }
#pragma unroll
        for (int f = 0; f < 2; ++f)
#pragma unroll
            for (int c = 0; c < 4; ++c)
                acc[c] = wmma_f16(afr[f], bfr[f][c], acc[c]);

        wait_async0();
        __syncthreads();
        buf = nb;
    }

#pragma unroll
    for (int c = 0; c < 4; ++c) {
        const int feat = colBase + wn * 64 + c * 16 + n0;
        const float bv = bias[feat];
#pragma unroll
        for (int r = 0; r < 8; ++r) {
            const int tok = rowBase + wm * 16 + r + 8 * half;
            out[(size_t)tok * DMODEL + feat] = acc[c][r] + bv;
        }
    }
}

// ---------------------------------------------------------------------------
// Flash attention (causal). Grid: (S/128, B*H). Block: 256 = 8 waves; wave w
// owns q rows [qb*128 + w*16, +16). 32-key KV tiles, online softmax.
// Keys interleaved even/odd across the two S fragments (key = j0 + 2*n0 + c)
// so the P tile is staged to LDS in ISA A-order with dword stores and read
// back as one contiguous 32B per-lane A-fragment (contraction index == key).
// ---------------------------------------------------------------------------
__global__ __launch_bounds__(256) void flash_attn(const _Float16* __restrict__ Qh,
                                                  const _Float16* __restrict__ Kh,
                                                  const _Float16* __restrict__ Vt,
                                                  _Float16* __restrict__ Oh) {
    __shared__ _Float16 Pl[8][16][40];  // per-wave P tile, A-order rows (80B pitch)
    const int tid  = threadIdx.x;
    const int lane = tid & 31;
    const int wave = tid >> 5;
    const int qb   = blockIdx.x;
    const int bh   = blockIdx.y;
    const int b    = bh >> 4;
    const int h    = bh & 15;
    const int n0   = lane & 15;
    const int half = lane >> 4;
    const int koff = half ? 8 : 0;
    const int qrow0 = qb * 128 + wave * 16;

    const int q2 = n0 >> 2;
    const int dw = ((((q2 & 1) << 1) | (q2 >> 1)) << 2) | (n0 & 3);

    const _Float16* Qbase = Qh + (size_t)bh * HTOK;
    const _Float16* Kbase = Kh + (size_t)bh * HTOK;
    const _Float16* Vbase = Vt + (size_t)bh * HTOK;

    v16h aq[2];
    {
        const _Float16* qr = Qbase + (size_t)(qrow0 + n0) * DEPTH;
#pragma unroll
        for (int f = 0; f < 2; ++f)
            aq[f] = cat16(*(const v8h*)(qr + f * 32 + half * 16),
                          *(const v8h*)(qr + f * 32 + half * 16 + 8));
    }

    float mrow[8], lrow[8];
    v8f acc[4];
#pragma unroll
    for (int r = 0; r < 8; ++r) { mrow[r] = -1e30f; lrow[r] = 0.0f; }
#pragma unroll
    for (int d = 0; d < 4; ++d) acc[d] = (v8f){0, 0, 0, 0, 0, 0, 0, 0};

    const int ntiles = qb * 4 + (wave >> 1) + 1;
    for (int jt = 0; jt < ntiles; ++jt) {
        const int j0  = jt * 32;
        const int key = j0 + 2 * n0;

        if (jt + 1 < ntiles) {
            __builtin_prefetch(Kbase + (size_t)(key + 32) * DEPTH, 0, 0);
            __builtin_prefetch(Vbase + (size_t)n0 * SEQ + j0 + 32, 0, 0);
        }

        v8f s[2];
#pragma unroll
        for (int c = 0; c < 2; ++c) {
            s[c] = (v8f){0, 0, 0, 0, 0, 0, 0, 0};
#pragma unroll
            for (int f = 0; f < 2; ++f) {
                const v16h bk = *(const v16h*)(Kbase +
                    (size_t)(key + c) * DEPTH + f * 32 + half * 16);
                s[c] = wmma_f16(aq[f], bk, s[c]);
            }
        }

        // Hoist V fragments; their latency hides behind the softmax below.
        v16h bv[4];
#pragma unroll
        for (int d = 0; d < 4; ++d)
            bv[d] = cat16(*(const v8h*)(Vbase + (size_t)(d * 16 + n0) * SEQ + j0 + koff), *(const v8h*)(Vbase + (size_t)(d * 16 + n0) * SEQ + j0 + 16 + koff));

        const bool needmask = (j0 + 31) > qrow0;

#pragma unroll
        for (int r = 0; r < 8; ++r) {
            const int qrow = qrow0 + r + 8 * half;
            float s0 = s[0][r], s1 = s[1][r];
            if (needmask) {
                if (key > qrow)     s0 = -1e30f;
                if (key + 1 > qrow) s1 = -1e30f;
            }
            float mx = fmaxf(s0, s1);
#pragma unroll
            for (int off = 8; off; off >>= 1) mx = fmaxf(mx, __shfl_xor(mx, off, 16));
            const float mnew  = fmaxf(mrow[r], mx);
            const float alpha = __expf(mrow[r] - mnew);
            const float p0 = __expf(s0 - mnew);
            const float p1 = __expf(s1 - mnew);
            float sum = p0 + p1;
#pragma unroll
            for (int off = 8; off; off >>= 1) sum += __shfl_xor(sum, off, 16);
            lrow[r] = lrow[r] * alpha + sum;
            mrow[r] = mnew;
#pragma unroll
            for (int d = 0; d < 4; ++d) acc[d][r] *= alpha;
            v2h pp = {(_Float16)p0, (_Float16)p1};
            *((v2h*)&Pl[wave][r + 8 * half][0] + dw) = pp;
        }
        asm volatile("s_wait_dscnt 0" ::: "memory");

        const _Float16* pr = &Pl[wave][n0][half * 16];
        const v16h ap = cat16(*(const v8h*)pr, *(const v8h*)(pr + 8));

#pragma unroll
        for (int d = 0; d < 4; ++d)
            acc[d] = wmma_f16(ap, bv[d], acc[d]);
    }

#pragma unroll
    for (int r = 0; r < 8; ++r) {
        const float inv = 1.0f / lrow[r];
        const int qrow = qrow0 + r + 8 * half;
        const size_t obase = ((size_t)b * SEQ + qrow) * DMODEL + h * DEPTH;
#pragma unroll
        for (int d = 0; d < 4; ++d)
            Oh[obase + d * 16 + n0] = (_Float16)(acc[d][r] * inv);
    }
}

// ---------------------------------------------------------------------------
extern "C" void kernel_launch(void* const* d_in, const int* in_sizes, int n_in,
                              void* d_out, int out_size, void* d_ws, size_t ws_size,
                              hipStream_t stream) {
    const float* v    = (const float*)d_in[0];
    const float* k    = (const float*)d_in[1];
    const float* q    = (const float*)d_in[2];
    // d_in[3] = mask (causality computed analytically)
    const float* wq_w = (const float*)d_in[4];
    const float* wq_b = (const float*)d_in[5];
    const float* wk_w = (const float*)d_in[6];
    const float* wk_b = (const float*)d_in[7];
    const float* wv_w = (const float*)d_in[8];
    const float* wv_b = (const float*)d_in[9];
    const float* wo_w = (const float*)d_in[10];
    const float* wo_b = (const float*)d_in[11];
    float* out = (float*)d_out;

    const size_t plane = (size_t)NTOK * DMODEL;   // 8M halves
    const size_t wsz   = (size_t)DMODEL * DMODEL; // 1M halves
    _Float16* Qh  = (_Float16*)d_ws;
    _Float16* Kh  = Qh + plane;
    _Float16* Vt  = Kh + plane;
    _Float16* Oh  = Vt + plane;
    _Float16* WtQ = Oh + plane;
    _Float16* WtK = WtQ + wsz;
    _Float16* WtV = WtK + wsz;
    _Float16* WtO = WtV + wsz;

    dim3 gT(DMODEL / 32, DMODEL / 32);  // 32 x 32
    w_transpose_f16<<<gT, 256, 0, stream>>>(wq_w, WtQ);
    w_transpose_f16<<<gT, 256, 0, stream>>>(wk_w, WtK);
    w_transpose_f16<<<gT, 256, 0, stream>>>(wv_w, WtV);
    w_transpose_f16<<<gT, 256, 0, stream>>>(wo_w, WtO);

    dim3 gGemm(NTOK / 64, DMODEL / 128);  // 128 x 8
    proj_qkv<0><<<gGemm, 256, 0, stream>>>(q, WtQ, wq_b, Qh);
    proj_qkv<1><<<gGemm, 256, 0, stream>>>(k, WtK, wk_b, Kh);
    proj_qkv<2><<<gGemm, 256, 0, stream>>>(v, WtV, wv_b, Vt);

    dim3 gAttn(SEQ / 128, BATCH * NHEADS);  // 16 x 64
    flash_attn<<<gAttn, 256, 0, stream>>>(Qh, Kh, Vt, Oh);

    proj_out<<<gGemm, 256, 0, stream>>>(Oh, WtO, wo_b, out);
}
